// DAMMFNDMODEL_75574244540856
// MI455X (gfx1250) — compile-verified, matrix operand order checked by reading, not run on hardware
//
#include <hip/hip_runtime.h>
#include <hip/hip_bf16.h>

// ---- Tensor Data Mover availability / arity detection -----------------------
#if defined(__has_builtin)
#if __has_builtin(__builtin_amdgcn_tensor_load_to_lds) && \
    __has_builtin(__builtin_amdgcn_s_wait_tensorcnt)
#define HAVE_TDM 1
#endif
#endif
#ifndef HAVE_TDM
#define HAVE_TDM 0
#endif

#if __has_include(<hip/amd_detail/amd_gfx1250_TDM.h>)
#define TDM_SIX_ARGS 1   // therock-10.0 headers -> clang-23 6-arg builtin
#else
#define TDM_SIX_ARGS 0   // ROCm 7.2 clang-22 5-arg builtin
#endif

typedef _Float16 v16h __attribute__((ext_vector_type(16)));
typedef _Float16 v8h  __attribute__((ext_vector_type(8)));
typedef float    v8f  __attribute__((ext_vector_type(8)));
typedef unsigned int u32x4 __attribute__((ext_vector_type(4)));
typedef int          i32x4 __attribute__((ext_vector_type(4)));
typedef int          i32x8 __attribute__((ext_vector_type(8)));

constexpr int BATCH   = 16384;
constexpr int DIM     = 1024;
constexpr int NHEAD   = 16;
constexpr int HEADDIM = 64;
constexpr int FFN     = 4096;
constexpr float LN_EPS = 1e-5f;

constexpr int EPI_F16      = 0;  // out16 = f16(acc + bias)
constexpr int EPI_F16_ADD  = 1;  // out16 = f16(acc + bias + add[m,n])
constexpr int EPI_F16_RELU = 2;  // out16 = f16(max(acc + bias, 0))
constexpr int EPI_F32      = 3;  // out32 = acc + bias

// ---------------------------------------------------------------------------
// TDM: DMA one [rows x 32] f16 tile (row stride K elems) from global into LDS,
// with 16B of LDS padding after every 64B row (matches the BK+8 LDS layout).
// D# per CDNA5 ISA.
// ---------------------------------------------------------------------------
#if HAVE_TDM
__device__ __forceinline__ void tdm_load_tile_f16(unsigned lds_byte_off,
                                                  const _Float16* gsrc,
                                                  int tileRows, int K,
                                                  int tensorRows) {
  unsigned long long ga = (unsigned long long)(uintptr_t)gsrc;
  u32x4 g0;
  g0[0] = 1u;                                       // count=1, user mode
  g0[1] = lds_byte_off;                             // lds_addr
  g0[2] = (unsigned)ga;                             // global_addr[31:0]
  g0[3] = (unsigned)((ga >> 32) & 0x01ffffffull)    // global_addr[56:32]
        | 0x80000000u;                              // type=2 ("image")
  i32x8 g1;
  g1[0] = (1 << 16)      // data_size = 2 bytes
        | (1 << 20)      // pad_enable
        | (3 << 22)      // pad_interval: every 16 DWORDs (=64B row)
        | (3 << 25);     // pad_amount: 4 DWORDs (=16B)
  g1[1] = (int)((unsigned)(K & 0xffff) << 16);                       // tensor_dim0 lo
  g1[2] = (int)(((unsigned)K >> 16) | ((unsigned)(tensorRows & 0xffff) << 16));
  g1[3] = (int)(((unsigned)tensorRows >> 16) | (32u << 16));         // tile_dim0=32
  g1[4] = tileRows & 0xffff;                                         // tile_dim1
  g1[5] = K;                                                         // dim0 stride lo
  g1[6] = 0;
  g1[7] = 0;
  i32x4 z4 = {0, 0, 0, 0};
#if TDM_SIX_ARGS
  i32x8 z8 = {0, 0, 0, 0, 0, 0, 0, 0};
  __builtin_amdgcn_tensor_load_to_lds(g0, g1, z4, z4, z8, 0);
#else
  __builtin_amdgcn_tensor_load_to_lds(g0, g1, z4, z4, 0);
#endif
}
#endif

// ---------------------------------------------------------------------------
// Prep kernels
// ---------------------------------------------------------------------------
// Weight convert + transpose: Wt[n*K + k] = (f16) W[k*N + n]
__global__ void k_cvt_wT(const float* __restrict__ W, _Float16* __restrict__ Wt,
                         int K, int N) {
  int id = blockIdx.x * blockDim.x + threadIdx.x;
  if (id >= K * N) return;
  int k = id / N;
  int n = id - k * N;
  Wt[(size_t)n * K + k] = (_Float16)W[id];
}

// f32 -> f16 activation conversion (vectorized 4-wide)
__global__ void k_cvt16(const float* __restrict__ in, _Float16* __restrict__ out) {
  size_t id = ((size_t)blockIdx.x * blockDim.x + threadIdx.x) * 4;
  float4 f = *(const float4*)&in[id];
  v8h dummy;
  _Float16 h0 = (_Float16)f.x, h1 = (_Float16)f.y,
           h2 = (_Float16)f.z, h3 = (_Float16)f.w;
  (void)dummy;
  _Float16* o = out + id;
  o[0] = h0; o[1] = h1; o[2] = h2; o[3] = h3;
}

// mean of 3 modalities, fused with f16 conversion
__global__ void k_mean3_16(const float* __restrict__ a, const float* __restrict__ b,
                           const float* __restrict__ c, _Float16* __restrict__ out) {
  size_t id = ((size_t)blockIdx.x * blockDim.x + threadIdx.x) * 4;
  float4 fa = *(const float4*)&a[id];
  float4 fb = *(const float4*)&b[id];
  float4 fc = *(const float4*)&c[id];
  const float s = 1.0f / 3.0f;
  _Float16* o = out + id;
  o[0] = (_Float16)((fa.x + fb.x + fc.x) * s);
  o[1] = (_Float16)((fa.y + fb.y + fc.y) * s);
  o[2] = (_Float16)((fa.z + fb.z + fc.z) * s);
  o[3] = (_Float16)((fa.w + fb.w + fc.w) * s);
}

// ---------------------------------------------------------------------------
// WMMA GEMM:  C[M,N] = A[M,K] @ Bt[N,K]^T  (+ epilogue)
// Pure-TDM double-buffered pipeline, one barrier per K-step:
//   wave0: s_wait_tensorcnt 0 -> barrier -> tensor_load_to_lds(next tiles)
//   all:   ds_load_b128 fragments -> v_wmma_f32_16x16x32_f16
// 256 threads = 8 wave32; wave grid 2(M) x 4(N); each wave: 4x2 tiles (64x32),
// f32 accumulators. Per-thread staging kept only as no-TDM compile fallback.
// ---------------------------------------------------------------------------
template<int EPI>
__global__ __launch_bounds__(256)
void k_wmma_gemm(const _Float16* __restrict__ A16, const _Float16* __restrict__ Bt,
                 const float* __restrict__ bias, const float* __restrict__ add,
                 _Float16* __restrict__ out16, float* __restrict__ out32,
                 int K, int N) {
  constexpr int BM = 128, BN = 128, BK = 32, LDT = BK + 8;  // 80B row stride
  __shared__ __align__(16) _Float16 As[2][BM][LDT];
  __shared__ __align__(16) _Float16 Bs[2][BN][LDT];

  const int tid  = threadIdx.x;
  const int bm   = blockIdx.y * BM;
  const int bn   = blockIdx.x * BN;
  const int wave = tid >> 5;
  const int lane = tid & 31;
  const int g    = lane >> 4;        // half-wave -> K sub-range select
  const int ml   = lane & 15;        // row within 16x16 tile
  const int wm   = (wave & 1) * 64;  // wave offset along M
  const int wn   = (wave >> 1) * 32; // wave offset along N
  const int nk = K >> 5;

#if !HAVE_TDM
  const int ldRow = tid >> 2;        // 0..63
  const int ldCol = (tid & 3) * 8;   // 0,8,16,24 halves
  uint4 arh[2], brh[2];              // staging regs (fallback only)
#endif

  v8f acc[4][2] = {};

  auto stage = [&](int buf, int k0) {
#if HAVE_TDM
    if (wave == 0) {
      tdm_load_tile_f16((unsigned)(uintptr_t)&As[buf][0][0],
                        A16 + (size_t)bm * K + k0, BM, K, BATCH);
      tdm_load_tile_f16((unsigned)(uintptr_t)&Bs[buf][0][0],
                        Bt + (size_t)bn * K + k0, BN, K, N);
    }
#else
#pragma unroll
    for (int p = 0; p < 2; ++p) {
      arh[p] = *(const uint4*)&A16[(size_t)(bm + p * 64 + ldRow) * K + k0 + ldCol];
      brh[p] = *(const uint4*)&Bt[(size_t)(bn + p * 64 + ldRow) * K + k0 + ldCol];
    }
#pragma unroll
    for (int p = 0; p < 2; ++p) {
      *(uint4*)&As[buf][p * 64 + ldRow][ldCol] = arh[p];
      *(uint4*)&Bs[buf][p * 64 + ldRow][ldCol] = brh[p];
    }
#endif
  };

  // ---- prologue: stage tile 0 into buffer 0 ----
  stage(0, 0);

  for (int t = 0; t < nk; ++t) {
    const int c = t & 1;
    const bool more = (t + 1) < nk;
    const int k1 = (t + 1) << 5;

#if HAVE_TDM
    if (wave == 0) __builtin_amdgcn_s_wait_tensorcnt(0);
#endif
    __syncthreads();                 // buffer c ready; buffer 1-c free
    if (more) stage(1 - c, k1);      // TDM(t+1) overlaps WMMA(t)

    // ---- fragments (ISA 16-bit A/B layout: half-wave g selects K runs
    //      [g*8, g*8+8) and [16+g*8, 16+g*8+8)) + WMMA ----
    v16h af[4], bf[2];
#pragma unroll
    for (int i = 0; i < 4; ++i) {
      const int row = wm + i * 16 + ml;
      union { v16h v; v8h h[2]; } u;
      u.h[0] = *(const v8h*)&As[c][row][g * 8];
      u.h[1] = *(const v8h*)&As[c][row][16 + g * 8];
      af[i] = u.v;
    }
#pragma unroll
    for (int j = 0; j < 2; ++j) {
      const int row = wn + j * 16 + ml;
      union { v16h v; v8h h[2]; } u;
      u.h[0] = *(const v8h*)&Bs[c][row][g * 8];
      u.h[1] = *(const v8h*)&Bs[c][row][16 + g * 8];
      bf[j] = u.v;
    }
#pragma unroll
    for (int i = 0; i < 4; ++i)
#pragma unroll
      for (int j = 0; j < 2; ++j)
        acc[i][j] = __builtin_amdgcn_wmma_f32_16x16x32_f16(
            false, af[i], false, bf[j], (short)0, acc[i][j], false, false);
  }

  // ---- epilogue: C/D layout -> m_local = g*8 + r, n_local = ml ----
#pragma unroll
  for (int i = 0; i < 4; ++i) {
#pragma unroll
    for (int j = 0; j < 2; ++j) {
      const int n     = bn + wn + j * 16 + ml;
      const int mbase = bm + wm + i * 16 + g * 8;
      const float bv  = bias[n];
#pragma unroll
      for (int r = 0; r < 8; ++r) {
        const size_t m = (size_t)(mbase + r);
        float val = acc[i][j][r] + bv;
        if constexpr (EPI == EPI_F16_ADD)  val += add[m * N + n];
        if constexpr (EPI == EPI_F16_RELU) val = fmaxf(val, 0.0f);
        if constexpr (EPI == EPI_F32)      out32[m * N + n] = val;
        else                               out16[m * N + n] = (_Float16)val;
      }
    }
  }
}

// ---------------------------------------------------------------------------
// Per-row: 3-key attention + residual + LayerNorm1.
// 256 threads/row: 16 threads per head, 4 dims per thread.
// ---------------------------------------------------------------------------
__global__ __launch_bounds__(256)
void k_attn_ln1(const _Float16* __restrict__ q16, const _Float16* __restrict__ k16,
                const _Float16* __restrict__ v16, const float* __restrict__ dr,
                const float* __restrict__ g1, const float* __restrict__ beta1,
                _Float16* __restrict__ x16, float* __restrict__ x32) {
  const int b    = blockIdx.x;
  const int tid  = threadIdx.x;
  const int lane = tid & 31;
  const int head = tid >> 4;
  const int l16  = tid & 15;
  const int col0 = head * HEADDIM + l16 * 4;
  const size_t row = (size_t)b * DIM;
  constexpr size_t MOD = (size_t)BATCH * DIM;

  float qv[4];
#pragma unroll
  for (int d = 0; d < 4; ++d) qv[d] = (float)q16[row + col0 + d];

  float s[3] = {0.f, 0.f, 0.f};
#pragma unroll
  for (int i = 0; i < 3; ++i) {
    const _Float16* kp = k16 + (size_t)i * MOD + row + col0;
#pragma unroll
    for (int d = 0; d < 4; ++d) s[i] += qv[d] * (float)kp[d];
  }
#pragma unroll
  for (int off = 8; off >= 1; off >>= 1)
#pragma unroll
    for (int i = 0; i < 3; ++i) s[i] += __shfl_xor(s[i], off, 32);

  const float scale = 0.125f;  // 1/sqrt(64)
  float mx = fmaxf(s[0] * scale, fmaxf(s[1] * scale, s[2] * scale));
  float e[3], esum = 0.f;
#pragma unroll
  for (int i = 0; i < 3; ++i) { e[i] = __expf(s[i] * scale - mx); esum += e[i]; }
  const float inv = 1.f / esum;

  float rv[4], ps = 0.f, pss = 0.f;
#pragma unroll
  for (int d = 0; d < 4; ++d) {
    float o = 0.f;
#pragma unroll
    for (int i = 0; i < 3; ++i)
      o += e[i] * inv * (float)v16[(size_t)i * MOD + row + col0 + d];
    float r = dr[row + col0 + d] + o;
    rv[d] = r; ps += r; pss += r * r;
  }

#pragma unroll
  for (int off = 16; off >= 1; off >>= 1) {
    ps  += __shfl_xor(ps, off, 32);
    pss += __shfl_xor(pss, off, 32);
  }
  __shared__ float red[16];
  __shared__ float smu, srs;
  if (lane == 0) { red[tid >> 5] = ps; red[8 + (tid >> 5)] = pss; }
  __syncthreads();
  if (tid == 0) {
    float S = 0.f, SS = 0.f;
    for (int w = 0; w < 8; ++w) { S += red[w]; SS += red[8 + w]; }
    float mu  = S / DIM;
    float var = SS / DIM - mu * mu;
    smu = mu; srs = rsqrtf(var + LN_EPS);
  }
  __syncthreads();
#pragma unroll
  for (int d = 0; d < 4; ++d) {
    const int c = col0 + d;
    float x = (rv[d] - smu) * srs * g1[c] + beta1[c];
    x16[row + c] = (_Float16)x;
    x32[row + c] = x;
  }
}

// ---------------------------------------------------------------------------
// Per-row: residual + LayerNorm2 + 3-way head + softmax -> out[B,3] f32
// ---------------------------------------------------------------------------
__global__ __launch_bounds__(256)
void k_ln2_head(const float* __restrict__ x32, const float* __restrict__ y32,
                const float* __restrict__ g2, const float* __restrict__ beta2,
                const float* __restrict__ Ww, const float* __restrict__ bw,
                float* __restrict__ out) {
  const int b    = blockIdx.x;
  const int tid  = threadIdx.x;
  const int lane = tid & 31;
  const size_t row = (size_t)b * DIM;
  const int col0 = tid * 4;

  float rv[4], ps = 0.f, pss = 0.f;
#pragma unroll
  for (int d = 0; d < 4; ++d) {
    float v = x32[row + col0 + d] + y32[row + col0 + d];
    rv[d] = v; ps += v; pss += v * v;
  }
#pragma unroll
  for (int off = 16; off >= 1; off >>= 1) {
    ps  += __shfl_xor(ps, off, 32);
    pss += __shfl_xor(pss, off, 32);
  }
  __shared__ float red[16];
  __shared__ float smu, srs;
  if (lane == 0) { red[tid >> 5] = ps; red[8 + (tid >> 5)] = pss; }
  __syncthreads();
  if (tid == 0) {
    float S = 0.f, SS = 0.f;
    for (int w = 0; w < 8; ++w) { S += red[w]; SS += red[8 + w]; }
    float mu  = S / DIM;
    float var = SS / DIM - mu * mu;
    smu = mu; srs = rsqrtf(var + LN_EPS);
  }
  __syncthreads();

  float lg[3] = {0.f, 0.f, 0.f};
#pragma unroll
  for (int d = 0; d < 4; ++d) {
    const int c = col0 + d;
    float x = (rv[d] - smu) * srs * g2[c] + beta2[c];
#pragma unroll
    for (int j = 0; j < 3; ++j) lg[j] += x * Ww[c * 3 + j];
  }
#pragma unroll
  for (int off = 16; off >= 1; off >>= 1)
#pragma unroll
    for (int j = 0; j < 3; ++j) lg[j] += __shfl_xor(lg[j], off, 32);

  __shared__ float lr[8][3];
  if (lane == 0)
    for (int j = 0; j < 3; ++j) lr[tid >> 5][j] = lg[j];
  __syncthreads();
  if (tid == 0) {
    float L[3];
    for (int j = 0; j < 3; ++j) {
      float s = bw[j];
      for (int w = 0; w < 8; ++w) s += lr[w][j];
      L[j] = s;
    }
    float mx = fmaxf(L[0], fmaxf(L[1], L[2]));
    float e0 = __expf(L[0] - mx), e1 = __expf(L[1] - mx), e2 = __expf(L[2] - mx);
    float inv = 1.f / (e0 + e1 + e2);
    out[(size_t)b * 3 + 0] = e0 * inv;
    out[(size_t)b * 3 + 1] = e1 * inv;
    out[(size_t)b * 3 + 2] = e2 * inv;
  }
}

// ---------------------------------------------------------------------------
extern "C" void kernel_launch(void* const* d_in, const int* in_sizes, int n_in,
                              void* d_out, int out_size, void* d_ws, size_t ws_size,
                              hipStream_t stream) {
  const float* m0    = (const float*)d_in[0];
  const float* m1    = (const float*)d_in[1];
  const float* m2    = (const float*)d_in[2];
  const float* dr    = (const float*)d_in[3];
  const float* Wg    = (const float*)d_in[4];
  const float* bg    = (const float*)d_in[5];
  const float* Wq    = (const float*)d_in[6];
  const float* bq    = (const float*)d_in[7];
  const float* Wk    = (const float*)d_in[8];
  const float* bk    = (const float*)d_in[9];
  const float* Wv    = (const float*)d_in[10];
  const float* bv    = (const float*)d_in[11];
  const float* W1    = (const float*)d_in[12];
  const float* b1    = (const float*)d_in[13];
  const float* W2    = (const float*)d_in[14];
  const float* b2    = (const float*)d_in[15];
  const float* g1    = (const float*)d_in[16];
  const float* beta1 = (const float*)d_in[17];
  const float* g2    = (const float*)d_in[18];
  const float* beta2 = (const float*)d_in[19];
  const float* Ww    = (const float*)d_in[20];
  const float* bw    = (const float*)d_in[21];

  char* p = (char*)d_ws;
  auto alloc = [&](size_t bytes) -> char* {
    char* r = p;
    p += (bytes + 255) & ~(size_t)255;
    return r;
  };
  const size_t BD = (size_t)BATCH * DIM;

  _Float16* WgT    = (_Float16*)alloc((size_t)DIM * DIM * 2);
  _Float16* WqT    = (_Float16*)alloc((size_t)DIM * DIM * 2);
  _Float16* WkT    = (_Float16*)alloc((size_t)DIM * DIM * 2);
  _Float16* WvT    = (_Float16*)alloc((size_t)DIM * DIM * 2);
  _Float16* W1T    = (_Float16*)alloc((size_t)DIM * FFN * 2);
  _Float16* W2T    = (_Float16*)alloc((size_t)FFN * DIM * 2);
  _Float16* mean16 = (_Float16*)alloc(BD * 2);
  _Float16* m16    = (_Float16*)alloc(3 * BD * 2);   // m0,m1,m2 as f16
  _Float16* drg16  = (_Float16*)alloc(BD * 2);
  _Float16* q16    = (_Float16*)alloc(BD * 2);
  _Float16* k16    = (_Float16*)alloc(3 * BD * 2);
  _Float16* v16    = (_Float16*)alloc(3 * BD * 2);
  _Float16* x16    = (_Float16*)alloc(BD * 2);
  float*    x32    = (float*)alloc(BD * 4);
  _Float16* h16    = (_Float16*)alloc((size_t)BATCH * FFN * 2);
  float*    y32    = (float*)alloc(BD * 4);

  // --- prep: weights (f16, transposed to [N][K]) + f16 activations ---
  k_cvt_wT<<<(DIM * DIM) / 256, 256, 0, stream>>>(Wg, WgT, DIM, DIM);
  k_cvt_wT<<<(DIM * DIM) / 256, 256, 0, stream>>>(Wq, WqT, DIM, DIM);
  k_cvt_wT<<<(DIM * DIM) / 256, 256, 0, stream>>>(Wk, WkT, DIM, DIM);
  k_cvt_wT<<<(DIM * DIM) / 256, 256, 0, stream>>>(Wv, WvT, DIM, DIM);
  k_cvt_wT<<<(DIM * FFN) / 256, 256, 0, stream>>>(W1, W1T, DIM, FFN);
  k_cvt_wT<<<(FFN * DIM) / 256, 256, 0, stream>>>(W2, W2T, FFN, DIM);
  k_mean3_16<<<BD / 1024, 256, 0, stream>>>(m0, m1, m2, mean16);
  k_cvt16<<<BD / 1024, 256, 0, stream>>>(m0, m16 + 0 * BD);
  k_cvt16<<<BD / 1024, 256, 0, stream>>>(m1, m16 + 1 * BD);
  k_cvt16<<<BD / 1024, 256, 0, stream>>>(m2, m16 + 2 * BD);

  const dim3 blk(256);
  const dim3 gD(DIM / 128, BATCH / 128);
  const dim3 gF(FFN / 128, BATCH / 128);

  // dr + mean@Wg + bg -> f16 (input to Wq GEMM)
  k_wmma_gemm<EPI_F16_ADD><<<gD, blk, 0, stream>>>(
      mean16, WgT, bg, dr, drg16, nullptr, DIM, DIM);
  // q
  k_wmma_gemm<EPI_F16><<<gD, blk, 0, stream>>>(
      drg16, WqT, bq, nullptr, q16, nullptr, DIM, DIM);
  // k_i, v_i
  for (int i = 0; i < 3; ++i) {
    k_wmma_gemm<EPI_F16><<<gD, blk, 0, stream>>>(
        m16 + (size_t)i * BD, WkT, bk, nullptr, k16 + (size_t)i * BD, nullptr,
        DIM, DIM);
    k_wmma_gemm<EPI_F16><<<gD, blk, 0, stream>>>(
        m16 + (size_t)i * BD, WvT, bv, nullptr, v16 + (size_t)i * BD, nullptr,
        DIM, DIM);
  }
  // attention + residual + LN1
  k_attn_ln1<<<BATCH, 256, 0, stream>>>(q16, k16, v16, dr, g1, beta1, x16, x32);
  // FFN up (relu) and down
  k_wmma_gemm<EPI_F16_RELU><<<gF, blk, 0, stream>>>(
      x16, W1T, b1, nullptr, h16, nullptr, DIM, FFN);
  k_wmma_gemm<EPI_F32><<<gD, blk, 0, stream>>>(
      h16, W2T, b2, nullptr, nullptr, y32, FFN, DIM);
  // residual + LN2 + head + softmax
  k_ln2_head<<<BATCH, 256, 0, stream>>>(x32, y32, g2, beta2, Ww, bw, (float*)d_out);
}
